// TruncationModule_69647189671968
// MI455X (gfx1250) — compile-verified, matrix operand order checked by reading, not run on hardware
//
#include <hip/hip_runtime.h>

typedef __attribute__((ext_vector_type(16))) __bf16 v16bf;
typedef __attribute__((ext_vector_type(8)))  float  v8f;

#define TBINS 1024
#define NROWS 32768

// WMMA 16-bit operand fragment pattern (ISA 7.12.2):
// v16bf element (2v+half) of lane-group g holds K = 16*(v>>2) + 2*(v&3) + 8g + half
__device__ __forceinline__ int frag_k(int v, int g) {
    return ((v >> 2) << 4) + ((v & 3) << 1) + (g << 3);
}

// WGP-scope prefetch (scope field 0: fill every cache level on a miss).
// __builtin_prefetch only reaches SYS/SE scope, which skips the WGP cache.
__device__ __forceinline__ void prefetch_wgp(const void* p) {
    asm volatile("global_prefetch_b8 %0, off" :: "v"(p) : "memory");
}

// ---------------------------------------------------------------------------
// Kernel 0: convert W (f32, row-major [T,T]) into bf16 B-matrix fragments.
// Fragment f = colTile*32 + ks covers columns [colTile*16,+16), k [ks*32,+32).
// Stored lane-major (32 lanes x 16 bf16 = 512 bf16 per fragment) so the GEMM
// kernel loads each lane's operand with contiguous 32-byte vector loads.
// Also zeroes the loss accumulator slot.
// ---------------------------------------------------------------------------
__global__ void __launch_bounds__(256)
wswizzle_kernel(const float* __restrict__ W, __bf16* __restrict__ wfrag,
                float* __restrict__ lossSlot) {
    if (blockIdx.x == 0 && threadIdx.x == 0) *lossSlot = 0.0f;
    const int wave = threadIdx.x >> 5, lane = threadIdx.x & 31;
    const int f = blockIdx.x * 8 + wave;     // 64 col-tiles * 32 k-steps = 2048 frags
    const int colTile = f >> 5, ks = f & 31;
    const int n = colTile * 16 + (lane & 15);   // output-logit column j
    const int g = lane >> 4;
    v16bf vec;
#pragma unroll
    for (int v = 0; v < 8; ++v) {
        // B[k][n] = W[n*T + k]  (logits = hazard @ W.T)
        const int k = ks * 32 + frag_k(v, g);
        float2 two = *(const float2*)(W + (size_t)n * TBINS + k);
        vec[2 * v]     = (__bf16)two.x;
        vec[2 * v + 1] = (__bf16)two.y;
    }
    ((v16bf*)wfrag)[(size_t)f * 32 + lane] = vec;
}

// ---------------------------------------------------------------------------
// Kernel 1: fused GEMM (bf16 WMMA, 4 N-tiles register-blocked per wave) +
// bias + argmax (sigmoid is monotone) + trunc_pos write + MSE-loss accumulate.
// One block = 16 rows x all 1024 cols; 2 passes x 8 waves x 4 tiles = 64 tiles.
// Each A fragment (2x ds_load_b128) feeds 4 WMMAs.
// ---------------------------------------------------------------------------
__global__ void __launch_bounds__(256)
gemm_argmax_kernel(const float* __restrict__ hz, const float* __restrict__ labels,
                   const float* __restrict__ bias, const __bf16* __restrict__ wfrag,
                   int* __restrict__ truncPos, float* __restrict__ lossSlot) {
    __shared__ __attribute__((aligned(32))) __bf16 aLds[16 * TBINS]; // 32 KB A panel, fragment order
    __shared__ float sVal[8][16];
    __shared__ int   sIdx[8][16];

    const int tid  = threadIdx.x;
    const int wave = tid >> 5, lane = tid & 31;
    const int r0   = blockIdx.x * 16;

    // Stage A (16 rows x 1024 K) into LDS as bf16 WMMA fragments (32 ksteps x 32 lane-slots)
    for (int s = tid; s < 1024; s += 256) {
        const int ks = s >> 5, ln = s & 31;
        const int m = ln & 15, g = ln >> 4;
        const float* src = hz + (size_t)(r0 + m) * TBINS + ks * 32;
        v16bf vec;
#pragma unroll
        for (int v = 0; v < 8; ++v) {
            float2 two = *(const float2*)(src + frag_k(v, g));
            vec[2 * v]     = (__bf16)two.x;
            vec[2 * v + 1] = (__bf16)two.y;
        }
        ((v16bf*)aLds)[s] = vec;
    }
    __syncthreads();

    float bestv[8];
    int   besti[8];
#pragma unroll
    for (int r = 0; r < 8; ++r) { bestv[r] = -3.4e38f; besti[r] = 0; }

    const v16bf* aP = (const v16bf*)aLds;
    for (int pass = 0; pass < 2; ++pass) {
        const int colTile = pass * 32 + wave * 4;   // this wave's 4 adjacent 16-col tiles
        const v16bf* bP = (const v16bf*)wfrag + (size_t)colTile * 1024;

        // Warm the 4 B streams into the WGP cache (per-lane addr -> whole wave
        // covers each full 1 KB fragment's cachelines).
        prefetch_wgp((const void*)(bP + lane));
        prefetch_wgp((const void*)(bP + 1024 + lane));
        prefetch_wgp((const void*)(bP + 2048 + lane));
        prefetch_wgp((const void*)(bP + 3072 + lane));

        v8f c0 = {};
        v8f c1 = {};
        v8f c2 = {};
        v8f c3 = {};
#pragma unroll 2
        for (int ks = 0; ks < 32; ++ks) {
            v16bf a  = aP[ks * 32 + lane];           // ds_load_b128 x2, reused 4x
            v16bf b0 = bP[ks * 32 + lane];           // global_load_b128 x2 (L2-hot)
            v16bf b1 = bP[1024 + ks * 32 + lane];
            v16bf b2 = bP[2048 + ks * 32 + lane];
            v16bf b3 = bP[3072 + ks * 32 + lane];
            c0 = __builtin_amdgcn_wmma_f32_16x16x32_bf16(
                     false, a, false, b0, (short)0, c0, false, false);
            c1 = __builtin_amdgcn_wmma_f32_16x16x32_bf16(
                     false, a, false, b1, (short)0, c1, false, false);
            c2 = __builtin_amdgcn_wmma_f32_16x16x32_bf16(
                     false, a, false, b2, (short)0, c2, false, false);
            c3 = __builtin_amdgcn_wmma_f32_16x16x32_bf16(
                     false, a, false, b3, (short)0, c3, false, false);
        }
        // D layout: lane holds column n = lane&15; VGPR r holds row m = r + 8*(lane>>4)
        const int   j0  = colTile * 16 + (lane & 15);
        const float bj0 = bias[j0];
        const float bj1 = bias[j0 + 16];
        const float bj2 = bias[j0 + 32];
        const float bj3 = bias[j0 + 48];
#pragma unroll
        for (int r = 0; r < 8; ++r) {
            // per-lane column index strictly increases (c0<c1<c2<c3 < next pass),
            // so strict '>' preserves first-occurrence argmax semantics
            const float v0 = c0[r] + bj0;    // sigmoid monotone -> argmax over raw logits
            if (v0 > bestv[r]) { bestv[r] = v0; besti[r] = j0; }
            const float v1 = c1[r] + bj1;
            if (v1 > bestv[r]) { bestv[r] = v1; besti[r] = j0 + 16; }
            const float v2 = c2[r] + bj2;
            if (v2 > bestv[r]) { bestv[r] = v2; besti[r] = j0 + 32; }
            const float v3 = c3[r] + bj3;
            if (v3 > bestv[r]) { bestv[r] = v3; besti[r] = j0 + 48; }
        }
    }

    // Reduce across the 16 lanes of each half-wave (columns), first-index on ties
#pragma unroll
    for (int r = 0; r < 8; ++r) {
        for (int off = 8; off >= 1; off >>= 1) {
            float ov = __shfl_xor(bestv[r], off, 32);
            int   oi = __shfl_xor(besti[r], off, 32);
            if (ov > bestv[r] || (ov == bestv[r] && oi < besti[r])) {
                bestv[r] = ov; besti[r] = oi;
            }
        }
    }
    if ((lane & 15) == 0) {
        const int base = (lane >> 4) * 8;     // rows 0-7 from half 0, 8-15 from half 1
#pragma unroll
        for (int r = 0; r < 8; ++r) { sVal[wave][base + r] = bestv[r]; sIdx[wave][base + r] = besti[r]; }
    }
    __syncthreads();

    // Reduce across the 8 waves (column ranges), write trunc_pos + loss
    if (tid < 16) {
        float bv = sVal[0][tid];
        int   bi = sIdx[0][tid];
#pragma unroll
        for (int w = 1; w < 8; ++w) {
            const float ov = sVal[w][tid];
            const int   oi = sIdx[w][tid];
            if (ov > bv || (ov == bv && oi < bi)) { bv = ov; bi = oi; }
        }
        const int row = r0 + tid;
        truncPos[row] = bi;
        const float d = (float)bi - labels[row];
        atomicAdd(lossSlot, d * d * (1.0f / (float)NROWS));   // global_atomic_add_f32
    }
}

// ---------------------------------------------------------------------------
// Kernel 2: survival curve S = cumprod(1 - h) with truncation decay.
// One wave per row: 32 elements per lane, chunk products + wave-scan.
// ---------------------------------------------------------------------------
__global__ void __launch_bounds__(256)
survival_kernel(const float* __restrict__ hz, const int* __restrict__ truncPos,
                float* __restrict__ S) {
    const int wave = threadIdx.x >> 5, lane = threadIdx.x & 31;
    const int row  = blockIdx.x * 8 + wave;
    const float* src = hz + (size_t)row * TBINS + lane * 32;

    float loc[32];
    float prod = 1.0f;
#pragma unroll
    for (int i = 0; i < 32; i += 4) {
        float4 h = *(const float4*)(src + i);
        prod *= (1.0f - h.x); loc[i]     = prod;
        prod *= (1.0f - h.y); loc[i + 1] = prod;
        prod *= (1.0f - h.z); loc[i + 2] = prod;
        prod *= (1.0f - h.w); loc[i + 3] = prod;
    }
    // inclusive scan of chunk totals across 32 lanes -> exclusive prefix
    float scan = prod;
#pragma unroll
    for (int d = 1; d < 32; d <<= 1) {
        float o = __shfl_up(scan, d, 32);
        if (lane >= d) scan *= o;
    }
    float excl = __shfl_up(scan, 1, 32);
    if (lane == 0) excl = 1.0f;

    const int p  = truncPos[row];
    const int t0 = lane * 32;
    float* dst = S + (size_t)row * TBINS + t0;
#pragma unroll
    for (int i = 0; i < 32; i += 4) {
        const int t = t0 + i;
        float v0 = excl * loc[i];
        float v1 = excl * loc[i + 1];
        float v2 = excl * loc[i + 2];
        float v3 = excl * loc[i + 3];
        float4 o;
        o.x = (t     >= p) ? v0 * __expf((float)(p - t))       : v0;
        o.y = (t + 1 >= p) ? v1 * __expf((float)(p - (t + 1))) : v1;
        o.z = (t + 2 >= p) ? v2 * __expf((float)(p - (t + 2))) : v2;
        o.w = (t + 3 >= p) ? v3 * __expf((float)(p - (t + 3))) : v3;
        *(float4*)(dst + i) = o;
    }
}

// ---------------------------------------------------------------------------
extern "C" void kernel_launch(void* const* d_in, const int* in_sizes, int n_in,
                              void* d_out, int out_size, void* d_ws, size_t ws_size,
                              hipStream_t stream) {
    (void)in_sizes; (void)n_in; (void)out_size; (void)ws_size;
    const float* hz     = (const float*)d_in[0];   // hazard_values [B,T] f32
    const float* labels = (const float*)d_in[1];   // labels_Y [B] f32
    const float* W      = (const float*)d_in[2];   // W [T,T] f32
    const float* bias   = (const float*)d_in[3];   // b [T] f32

    // d_out layout (return order): trunc_pos [B] int32, S [B,T] f32, loss [1] f32
    int*   truncPos = (int*)d_out;
    float* outF     = (float*)d_out;
    float* Sout     = outF + NROWS;
    float* lossSlot = outF + (size_t)NROWS + (size_t)NROWS * (size_t)TBINS;

    __bf16* wfrag = (__bf16*)d_ws;                 // 2 MB swizzled bf16 W (L2-resident)

    wswizzle_kernel   <<<256,        256, 0, stream>>>(W, wfrag, lossSlot);
    gemm_argmax_kernel<<<NROWS / 16, 256, 0, stream>>>(hz, labels, bias, wfrag,
                                                       truncPos, lossSlot);
    survival_kernel   <<<NROWS / 8,  256, 0, stream>>>(hz, truncPos, Sout);
}
